// ConvCRF_60644938219645
// MI455X (gfx1250) — matrix operand order repaired and checked by reading, not compiled, not run on hardware
//
#include <hip/hip_runtime.h>

// Problem dims (compile-time)
#define B_ 32
#define T_ 4096
#define F_ 768
#define H_ 512
#define L_ 3
#define K_ 5
#define S_ 2

typedef __bf16 bf16;
typedef __attribute__((ext_vector_type(16))) __bf16 v16bf;
typedef __attribute__((ext_vector_type(8)))  __bf16 v8bf;
typedef __attribute__((ext_vector_type(8)))  float  v8f;

static __device__ __forceinline__ v8f wmma_bf16(v16bf a, v16bf b, v8f c) {
  // D = A(16x32 bf16) * B(32x16 bf16) + C(16x16 f32)
  return __builtin_amdgcn_wmma_f32_16x16x32_bf16(false, a, false, b, (short)0, c, false, false);
}

// gfx1250 async global->LDS DMA (ASYNCcnt-tracked), ISA cdna5 ch.10/15.18.
// VDST = per-lane LDS byte offset, VADDR = 64-bit global address.
static __device__ __forceinline__ void async_load_b128(void* lds, const void* g) {
  unsigned l = (unsigned)(uintptr_t)lds;           // low 32 bits == LDS offset
  unsigned long long a = (unsigned long long)(uintptr_t)g;
  asm volatile("global_load_async_to_lds_b128 %0, %1, off" :: "v"(l), "v"(a) : "memory");
}
static __device__ __forceinline__ void wait_asynccnt0() {
#if __has_builtin(__builtin_amdgcn_s_wait_asynccnt)
  __builtin_amdgcn_s_wait_asynccnt(0);
#else
  asm volatile("s_wait_asynccnt 0x0" ::: "memory");
#endif
}

// ---------------------------------------------------------------------------
// Weight packing into WMMA B-fragment lane order, split into bf16 hi/lo.
// B tile is 32(K) x 16(N). Lane l (0..31): n = ntile*16 + (l&15),
// K = (l<16 ? 0..7 then 16..23 : 8..15 then 24..31), stored as 16 contiguous halfs per lane.
// ---------------------------------------------------------------------------

// w_in: [F=768, H=512] row-major.  KT=24 (768/32), NT=32 (512/16).
__global__ void pack_win_kernel(const float* __restrict__ w,
                                bf16* __restrict__ hi, bf16* __restrict__ lo) {
  int idx = blockIdx.x * 256 + threadIdx.x;           // 24*32*32*16 = 393216
  if (idx >= 24 * 32 * 32 * 16) return;
  int h    = idx & 15;
  int lane = (idx >> 4) & 31;
  int nt   = (idx >> 9) & 31;
  int kt   = idx >> 14;
  int n    = nt * 16 + (lane & 15);
  int kb   = (lane < 16) ? 0 : 16;
  int k    = kt * 32 + (kb >> 1) + (h & 7) + ((h & 8) << 1);
  float x  = w[(size_t)k * H_ + n];
  bf16 xh  = (bf16)x;
  hi[idx]  = xh;
  lo[idx]  = (bf16)(x - (float)xh);
}

// conv_w: [L, O=512, I=512, K=5]. GEMM K-dim = tap*512 + c (size 2560).
// KT=80 per layer, NT=32. Per-layer elems = 80*32*32*16 = 1310720.
__global__ void pack_conv_kernel(const float* __restrict__ w,
                                 bf16* __restrict__ hi, bf16* __restrict__ lo) {
  int idx = blockIdx.x * 256 + threadIdx.x;           // 3*1310720 = 3932160
  if (idx >= 3 * 80 * 32 * 32 * 16) return;
  int h    = idx & 15;
  int lane = (idx >> 4) & 31;
  int nt   = (idx >> 9) & 31;
  int kt   = (idx >> 14) % 80;
  int l    = idx / (80 << 14);
  int n    = nt * 16 + (lane & 15);                   // out channel
  int kb   = (lane < 16) ? 0 : 16;
  int kk   = kt * 32 + (kb >> 1) + (h & 7) + ((h & 8) << 1); // 0..2559
  int k    = kk >> 9;                                 // tap 0..4
  int c    = kk & (H_ - 1);                           // in channel
  float x  = w[(((size_t)l * H_ + n) * H_ + c) * K_ + k];
  bf16 xh  = (bf16)x;
  hi[idx]  = xh;
  lo[idx]  = (bf16)(x - (float)xh);
}

// ---------------------------------------------------------------------------
// A-fragment load from LDS (16x32 bf16 row-major chunk).
// Lane l: row = l&15; halfs 0..7 = K (kb..kb+7), halfs 8..15 = K (16+kb..),
// kb = (l<16 ? 0 : 8).
// ---------------------------------------------------------------------------
static __device__ __forceinline__ v16bf load_a_frag(const bf16* sh, int arow, int kb) {
  v8bf a0 = *(const v8bf*)&sh[arow * 32 + kb];
  v8bf a1 = *(const v8bf*)&sh[arow * 32 + 16 + kb];
  v16bf out;
#pragma unroll
  for (int i = 0; i < 8; ++i) { out[i] = a0[i]; out[i + 8] = a1[i]; }
  return out;
}

// ---------------------------------------------------------------------------
// Projection GEMM (131072x768 @ 768x512) + bias + fused LayerNorm.
// Block = 8 waves; block owns 16 rows x all 512 cols (wave w -> cols 64w..).
// Output written pre-split as bf16 hi/lo for the async conv stack.
// ---------------------------------------------------------------------------
__global__ __launch_bounds__(256)
void proj_ln_kernel(const float* __restrict__ X,
                    const bf16* __restrict__ Whi, const bf16* __restrict__ Wlo,
                    const float* __restrict__ b_in,
                    const float* __restrict__ ln_g, const float* __restrict__ ln_b,
                    bf16* __restrict__ Yhi, bf16* __restrict__ Ylo) {
  __shared__ bf16 shA_hi[16 * 32];
  __shared__ bf16 shA_lo[16 * 32];
  __shared__ float rsum[16];
  __shared__ float rsq[16];

  const int tid    = threadIdx.x;
  const int lane   = tid & 31;
  const int wid    = tid >> 5;
  const int lanelo = lane & 15;
  const int kb     = (lane < 16) ? 0 : 8;
  const int grow0  = blockIdx.x * 16;

  v8f acc[4];
#pragma unroll
  for (int j = 0; j < 4; ++j)
#pragma unroll
    for (int r = 0; r < 8; ++r) acc[j][r] = 0.f;

  for (int kt = 0; kt < 24; ++kt) {
    for (int e = tid; e < 512; e += 256) {
      int row = e >> 5, col = e & 31;
      float x = X[(size_t)(grow0 + row) * F_ + kt * 32 + col];
      bf16 xh = (bf16)x;
      shA_hi[e] = xh;
      shA_lo[e] = (bf16)(x - (float)xh);
    }
    __syncthreads();
    v16bf Ahi = load_a_frag(shA_hi, lanelo, kb);
    v16bf Alo = load_a_frag(shA_lo, lanelo, kb);
#pragma unroll
    for (int j = 0; j < 4; ++j) {
      int nt = wid * 4 + j;
      size_t bo = (((size_t)kt * 32 + nt) * 32 + lane) * 16;
      v16bf bh = *(const v16bf*)&Whi[bo];
      v16bf bl = *(const v16bf*)&Wlo[bo];
      acc[j] = wmma_bf16(Ahi, bh, acc[j]);
      acc[j] = wmma_bf16(Ahi, bl, acc[j]);
      acc[j] = wmma_bf16(Alo, bh, acc[j]);
    }
    __syncthreads();
  }

  // Epilogue: bias + LayerNorm over full 512-wide row (block covers it all).
  if (tid < 16) { rsum[tid] = 0.f; rsq[tid] = 0.f; }
  __syncthreads();

  float vals[4][8];
#pragma unroll
  for (int j = 0; j < 4; ++j) {
    int col = (wid * 4 + j) * 16 + lanelo;
    float bj = b_in[col];
#pragma unroll
    for (int r = 0; r < 8; ++r) vals[j][r] = acc[j][r] + bj;
  }
#pragma unroll
  for (int r = 0; r < 8; ++r) {
    float s = vals[0][r] + vals[1][r] + vals[2][r] + vals[3][r];
    float q = vals[0][r] * vals[0][r] + vals[1][r] * vals[1][r] +
              vals[2][r] * vals[2][r] + vals[3][r] * vals[3][r];
#pragma unroll
    for (int m = 1; m < 16; m <<= 1) { s += __shfl_xor(s, m); q += __shfl_xor(q, m); }
    if (lanelo == 0) {
      int rowm = r + ((lane < 16) ? 0 : 8);
      atomicAdd(&rsum[rowm], s);
      atomicAdd(&rsq[rowm], q);
    }
  }
  __syncthreads();
#pragma unroll
  for (int j = 0; j < 4; ++j) {
    int col = (wid * 4 + j) * 16 + lanelo;
    float g = ln_g[col], bb = ln_b[col];
#pragma unroll
    for (int r = 0; r < 8; ++r) {
      int rowm = r + ((lane < 16) ? 0 : 8);
      float mean = rsum[rowm] * (1.f / 512.f);
      float var  = rsq[rowm] * (1.f / 512.f) - mean * mean;
      float rstd = rsqrtf(var + 1e-5f);
      float y    = (vals[j][r] - mean) * rstd * g + bb;
      bf16 yh    = (bf16)y;
      size_t o   = (size_t)(grow0 + rowm) * H_ + col;
      Yhi[o] = yh;
      Ylo[o] = (bf16)(y - (float)yh);
    }
  }
}

// ---------------------------------------------------------------------------
// Conv1d(512->512, K=5, pad=2) as implicit GEMM (K-dim 2560) + bias+BN+ReLU.
// A-tile staged via GLOBAL_LOAD_ASYNC_TO_LDS_B128 with double-buffered LDS:
// k-step kt+1's DMA runs concurrently with kt's WMMAs.
// ---------------------------------------------------------------------------
__global__ __launch_bounds__(256)
void conv_bn_kernel(const bf16* __restrict__ Xhi, const bf16* __restrict__ Xlo,
                    const bf16* __restrict__ Whi, const bf16* __restrict__ Wlo,
                    const float* __restrict__ conv_b,
                    const float* __restrict__ bn_g, const float* __restrict__ bn_b,
                    const float* __restrict__ bn_mean, const float* __restrict__ bn_var,
                    bf16* __restrict__ Yhi, bf16* __restrict__ Ylo) {
  __shared__ bf16 shA[2][2][16 * 32];   // [buffer][hi/lo][row*32+col]

  const int tid    = threadIdx.x;
  const int lane   = tid & 31;
  const int wid    = tid >> 5;
  const int lanelo = lane & 15;
  const int kb     = (lane < 16) ? 0 : 8;
  const int grow0  = blockIdx.x * 16;
  const int b      = grow0 / T_;     // tile never straddles a batch (T%16==0)
  const int t0     = grow0 % T_;

  // Staging role: threads 0..127. 64 threads per (hi/lo) array:
  // thread u covers row=u>>2, 16-byte segment seg=u&3 of the 64-byte row.
  const int st_hl  = (tid >> 6) & 1;
  const int st_u   = tid & 63;
  const int st_row = st_u >> 2;
  const int st_seg = st_u & 3;
  const bf16* st_src_base = st_hl ? Xlo : Xhi;

  v8f acc[4];
#pragma unroll
  for (int j = 0; j < 4; ++j)
#pragma unroll
    for (int r = 0; r < 8; ++r) acc[j][r] = 0.f;

  // stage(kt, buf): DMA 16x32 hi+lo chunk for k-step kt into LDS buffer buf.
  auto stage = [&](int kt, int buf) {
    if (tid < 128) {
      int k  = kt >> 4;                  // tap 0..4
      int c0 = (kt & 15) * 32;           // in-channel chunk
      int tt = t0 + st_row + (k - 2);    // shifted time row (pad=2)
      bf16* dst = &shA[buf][st_hl][st_row * 32 + st_seg * 8];
      if (tt >= 0 && tt < T_) {
        const bf16* src = st_src_base + ((size_t)b * T_ + tt) * H_ + c0 + st_seg * 8;
        async_load_b128(dst, src);
      } else {
        unsigned* z = (unsigned*)dst;    // zero-pad boundary rows (16 B)
        z[0] = 0u; z[1] = 0u; z[2] = 0u; z[3] = 0u;
      }
    }
  };

  stage(0, 0);
  wait_asynccnt0();
  __syncthreads();

  for (int kt = 0; kt < 80; ++kt) {
    int cur = kt & 1;
    if (kt + 1 < 80) stage(kt + 1, cur ^ 1);   // overlap next DMA with WMMAs

    v16bf Ahi = load_a_frag(shA[cur][0], lanelo, kb);
    v16bf Alo = load_a_frag(shA[cur][1], lanelo, kb);
#pragma unroll
    for (int j = 0; j < 4; ++j) {
      int nt = wid * 4 + j;
      size_t bo = (((size_t)kt * 32 + nt) * 32 + lane) * 16;
      v16bf bh = *(const v16bf*)&Whi[bo];
      v16bf bl = *(const v16bf*)&Wlo[bo];
      acc[j] = wmma_bf16(Ahi, bh, acc[j]);
      acc[j] = wmma_bf16(Ahi, bl, acc[j]);
      acc[j] = wmma_bf16(Alo, bh, acc[j]);
    }
    wait_asynccnt0();
    __syncthreads();
  }

#pragma unroll
  for (int j = 0; j < 4; ++j) {
    int col = (wid * 4 + j) * 16 + lanelo;
    float cb = conv_b[col];
    float sc = bn_g[col] * rsqrtf(bn_var[col] + 1e-5f);
    float sh = bn_b[col] - bn_mean[col] * sc;
#pragma unroll
    for (int r = 0; r < 8; ++r) {
      int rowm = r + ((lane < 16) ? 0 : 8);
      float v  = fmaxf((acc[j][r] + cb) * sc + sh, 0.f);
      bf16 vh  = (bf16)v;
      size_t o = (size_t)(grow0 + rowm) * H_ + col;
      Yhi[o] = vh;
      Ylo[o] = (bf16)(v - (float)vh);
    }
  }
}

// ---------------------------------------------------------------------------
// Emission: [B*T,512] @ [512,2] + b_em. One wave per row, shfl reduction.
// Consumes hi/lo split activations (x = hi + lo).
// ---------------------------------------------------------------------------
__global__ __launch_bounds__(256)
void emission_kernel(const bf16* __restrict__ Xhi, const bf16* __restrict__ Xlo,
                     const float* __restrict__ w_em, const float* __restrict__ b_em,
                     float* __restrict__ em) {
  int lane = threadIdx.x & 31, wid = threadIdx.x >> 5;
  size_t row = (size_t)blockIdx.x * 8 + wid;
  float a0 = 0.f, a1 = 0.f;
#pragma unroll 4
  for (int i = 0; i < 16; ++i) {
    int h = lane + 32 * i;
    float x = (float)Xhi[row * H_ + h] + (float)Xlo[row * H_ + h];
    a0 += x * w_em[h * 2 + 0];
    a1 += x * w_em[h * 2 + 1];
  }
  for (int m = 1; m < 32; m <<= 1) { a0 += __shfl_xor(a0, m); a1 += __shfl_xor(a1, m); }
  if (lane == 0) {
    em[row * 2 + 0] = a0 + b_em[0];
    em[row * 2 + 1] = a1 + b_em[1];
  }
}

// ---------------------------------------------------------------------------
// CRF NLL: one wave, lane = batch element (B=32 == wave32). S=2 unrolled.
// ---------------------------------------------------------------------------
__global__ void crf_nll_kernel(const float* __restrict__ em, const int* __restrict__ labels,
                               const float* __restrict__ start, const float* __restrict__ endp,
                               const float* __restrict__ trans, float* __restrict__ loss_out) {
  int bb = threadIdx.x;
  float t00 = trans[0], t01 = trans[1], t10 = trans[2], t11 = trans[3];
  size_t base = (size_t)bb * T_;
  float s0 = start[0] + em[base * 2 + 0];
  float s1 = start[1] + em[base * 2 + 1];
  int prev = labels[base];
  float num = start[prev] + em[base * 2 + prev];
  for (int t = 1; t < T_; ++t) {
    size_t o = (base + t) * 2;
    float e0 = em[o], e1 = em[o + 1];
    float a = s0 + t00, c = s1 + t10;
    float m0 = fmaxf(a, c);
    float n0 = m0 + logf(expf(a - m0) + expf(c - m0)) + e0;
    float a2 = s0 + t01, c2 = s1 + t11;
    float m1 = fmaxf(a2, c2);
    float n1 = m1 + logf(expf(a2 - m1) + expf(c2 - m1)) + e1;
    s0 = n0; s1 = n1;
    int cur = labels[base + t];
    num += trans[prev * 2 + cur] + (cur ? e1 : e0);
    prev = cur;
  }
  num += endp[prev];
  float a = s0 + endp[0], c = s1 + endp[1];
  float m = fmaxf(a, c);
  float den = m + logf(expf(a - m) + expf(c - m));
  float part = den - num;          // loss contribution = -(num - den)
  for (int mm = 1; mm < 32; mm <<= 1) part += __shfl_xor(part, mm);
  if (bb == 0) *loss_out = part;
}

// ---------------------------------------------------------------------------
// Viterbi: forward pass with backpointers, then traceback. One wave.
// argmax tie-break: lowest index first (matches jnp.argmax).
// ---------------------------------------------------------------------------
__global__ void viterbi_kernel(const float* __restrict__ em,
                               const float* __restrict__ start, const float* __restrict__ endp,
                               const float* __restrict__ trans,
                               unsigned char* __restrict__ bp, float* __restrict__ pred) {
  int bb = threadIdx.x;
  float t00 = trans[0], t01 = trans[1], t10 = trans[2], t11 = trans[3];
  size_t base = (size_t)bb * T_;
  float v0 = start[0] + em[base * 2 + 0];
  float v1 = start[1] + em[base * 2 + 1];
  for (int t = 1; t < T_; ++t) {
    size_t o = (base + t) * 2;
    float e0 = em[o], e1 = em[o + 1];
    float a = v0 + t00, c = v1 + t10;
    unsigned char bp0 = (c > a) ? 1 : 0;
    float n0 = fmaxf(a, c) + e0;
    float a2 = v0 + t01, c2 = v1 + t11;
    unsigned char bp1 = (c2 > a2) ? 1 : 0;
    float n1 = fmaxf(a2, c2) + e1;
    size_t bo = ((size_t)t * B_ + bb) * 2;
    bp[bo] = bp0; bp[bo + 1] = bp1;
    v0 = n0; v1 = n1;
  }
  int tag = (v1 + endp[1] > v0 + endp[0]) ? 1 : 0;
  pred[base + T_ - 1] = (float)tag;
  for (int t = T_ - 1; t >= 1; --t) {
    tag = bp[((size_t)t * B_ + bb) * 2 + tag];
    pred[base + t - 1] = (float)tag;
  }
}

// ---------------------------------------------------------------------------
extern "C" void kernel_launch(void* const* d_in, const int* in_sizes, int n_in,
                              void* d_out, int out_size, void* d_ws, size_t ws_size,
                              hipStream_t stream) {
  (void)in_sizes; (void)n_in; (void)out_size; (void)ws_size;

  const float* features  = (const float*)d_in[0];
  const int*   labels    = (const int*)d_in[1];
  const float* w_in      = (const float*)d_in[2];
  const float* b_in      = (const float*)d_in[3];
  const float* ln_g      = (const float*)d_in[4];
  const float* ln_b      = (const float*)d_in[5];
  const float* conv_w    = (const float*)d_in[6];
  const float* conv_b    = (const float*)d_in[7];
  const float* bn_g      = (const float*)d_in[8];
  const float* bn_b      = (const float*)d_in[9];
  const float* bn_mean   = (const float*)d_in[10];
  const float* bn_var    = (const float*)d_in[11];
  const float* w_em      = (const float*)d_in[12];
  const float* b_em      = (const float*)d_in[13];
  const float* crf_start = (const float*)d_in[14];
  const float* crf_end   = (const float*)d_in[15];
  const float* crf_trans = (const float*)d_in[16];

  // Workspace layout (activations stored split as bf16 hi/lo)
  char* ws = (char*)d_ws;
  size_t off = 0;
  const size_t actElems = (size_t)B_ * T_ * H_;
  bf16* actAhi = (bf16*)(ws + off); off += actElems * sizeof(bf16);
  bf16* actAlo = (bf16*)(ws + off); off += actElems * sizeof(bf16);
  bf16* actBhi = (bf16*)(ws + off); off += actElems * sizeof(bf16);
  bf16* actBlo = (bf16*)(ws + off); off += actElems * sizeof(bf16);
  const size_t winElems  = (size_t)24 * 32 * 32 * 16;             // 393216
  bf16* win_hi = (bf16*)(ws + off); off += winElems * sizeof(bf16);
  bf16* win_lo = (bf16*)(ws + off); off += winElems * sizeof(bf16);
  const size_t convElemsL = (size_t)80 * 32 * 32 * 16;            // per layer
  const size_t convElems  = 3 * convElemsL;
  bf16* conv_hi = (bf16*)(ws + off); off += convElems * sizeof(bf16);
  bf16* conv_lo = (bf16*)(ws + off); off += convElems * sizeof(bf16);
  unsigned char* bp = (unsigned char*)(ws + off); off += (size_t)T_ * B_ * S_;

  // Output layout: predictions [B*T] (as float), emissions [B*T*S], loss [1]
  float* pred = (float*)d_out;
  float* em   = pred + (size_t)B_ * T_;
  float* loss = em + (size_t)B_ * T_ * S_;

  // 1) pack weights (bf16 hi/lo, WMMA B-fragment order)
  pack_win_kernel<<<(int)((winElems + 255) / 256), 256, 0, stream>>>(w_in, win_hi, win_lo);
  pack_conv_kernel<<<(int)((convElems + 255) / 256), 256, 0, stream>>>(conv_w, conv_hi, conv_lo);

  // 2) projection + LayerNorm -> actA (hi/lo)
  const int nBlocks = B_ * T_ / 16;   // 8192
  proj_ln_kernel<<<nBlocks, 256, 0, stream>>>(features, win_hi, win_lo,
                                              b_in, ln_g, ln_b, actAhi, actAlo);

  // 3) conv stack (ping-pong actA/actB), async-DMA staged
  conv_bn_kernel<<<nBlocks, 256, 0, stream>>>(actAhi, actAlo,
      conv_hi + 0 * convElemsL, conv_lo + 0 * convElemsL,
      conv_b + 0 * H_, bn_g + 0 * H_, bn_b + 0 * H_, bn_mean + 0 * H_, bn_var + 0 * H_,
      actBhi, actBlo);
  conv_bn_kernel<<<nBlocks, 256, 0, stream>>>(actBhi, actBlo,
      conv_hi + 1 * convElemsL, conv_lo + 1 * convElemsL,
      conv_b + 1 * H_, bn_g + 1 * H_, bn_b + 1 * H_, bn_mean + 1 * H_, bn_var + 1 * H_,
      actAhi, actAlo);
  conv_bn_kernel<<<nBlocks, 256, 0, stream>>>(actAhi, actAlo,
      conv_hi + 2 * convElemsL, conv_lo + 2 * convElemsL,
      conv_b + 2 * H_, bn_g + 2 * H_, bn_b + 2 * H_, bn_mean + 2 * H_, bn_var + 2 * H_,
      actBhi, actBlo);

  // 4) emissions -> d_out emissions segment
  emission_kernel<<<B_ * T_ / 8, 256, 0, stream>>>(actBhi, actBlo, w_em, b_em, em);

  // 5) CRF loss + Viterbi decode (one wave each; lane == batch element)
  crf_nll_kernel<<<1, 32, 0, stream>>>(em, labels, crf_start, crf_end, crf_trans, loss);
  viterbi_kernel<<<1, 32, 0, stream>>>(em, crf_start, crf_end, crf_trans, bp, pred);
}
